// ISFI_90202903151069
// MI455X (gfx1250) — matrix operand order repaired and checked by reading, not compiled, not run on hardware
//
#include <hip/hip_runtime.h>
#include <math.h>

// ---------- types for WMMA ----------
typedef __attribute__((ext_vector_type(16))) __bf16          v16bf;
typedef __attribute__((ext_vector_type(8)))  float           v8f;
typedef __attribute__((ext_vector_type(2)))  float           v2f;
typedef __attribute__((ext_vector_type(8)))  unsigned short  ush8;
typedef __attribute__((ext_vector_type(4)))  unsigned short  ush4;

struct U16x16 { ush8 lo; ush8 hi; };   // 32 bytes -> v16bf

#define HW   65536
#define Himg 256
#define Wimg 256
#define CC   64
#define BB   4
#define LSTR 40   // LDS row stride in ushorts (80B = 5*16B): 16B-aligned rows, conflict-free

// ---------- helpers ----------
static __device__ __forceinline__ unsigned short f2b_bits(float f) {
    unsigned int u = __float_as_uint(f);
    u += 0x7fffu + ((u >> 16) & 1u);      // round-to-nearest-even
    return (unsigned short)(u >> 16);
}
static __device__ __forceinline__ float gelu_exact(float x) {
    return 0.5f * x * (1.0f + erff(x * 0.70710678118654752440f));
}
static __device__ __forceinline__ v16bf frag16(const unsigned short* p0,
                                               const unsigned short* p1) {
    U16x16 u;
    u.lo = *(const ush8*)p0;
    u.hi = *(const ush8*)p1;
    return __builtin_bit_cast(v16bf, u);
}

// ---------- weight prepack: fp32 -> bf16 bits (flat; OIHW == [O][C*kh*kw]) ----------
__global__ __launch_bounds__(256) void prep_bf16(const float* __restrict__ s,
                                                 unsigned short* __restrict__ d, int n) {
    int i = blockIdx.x * 256 + threadIdx.x;
    if (i < n) d[i] = f2b_bits(s[i]);
}

// ---------- LayerNorm over channel dim (per pixel, C=64) ----------
__global__ __launch_bounds__(256) void layernorm_c_kernel(const float* __restrict__ in,
                                                          const float* __restrict__ w,
                                                          const float* __restrict__ bias,
                                                          float* __restrict__ out) {
    int gid = blockIdx.x * 256 + threadIdx.x;       // over B*HW
    int b = gid >> 16, p = gid & (HW - 1);
    const float* base = in + (size_t)b * CC * HW + p;
    float s1 = 0.f, s2 = 0.f;
    for (int c = 0; c < CC; ++c) { float v = base[c * HW]; s1 += v; s2 += v * v; }
    float mean = s1 * (1.f / CC);
    float var  = fmaxf(s2 * (1.f / CC) - mean * mean, 0.f);
    float rstd = rsqrtf(var + 1e-5f);
    float* ob = out + (size_t)b * CC * HW + p;
    for (int c = 0; c < CC; ++c) {
        float v = base[c * HW];
        ob[c * HW] = (v - mean) * rstd * w[c] + bias[c];
    }
}

// ---------- depthwise 3x3 conv (zero pad), Ct channels ----------
__global__ __launch_bounds__(256) void dwconv3x3(const float* __restrict__ in,
                                                 const float* __restrict__ wdw,
                                                 float* __restrict__ out, int Ct) {
    int gid = blockIdx.x * 256 + threadIdx.x;       // over B*Ct*HW
    int p  = gid & (HW - 1);
    int hc = gid >> 16;
    int c  = hc % Ct;
    int h = p >> 8, w_ = p & 255;
    const float* ib = in + ((size_t)hc) * HW;
    const float* wc = wdw + c * 9;
    float acc = 0.f;
    #pragma unroll
    for (int dy = -1; dy <= 1; ++dy) {
        int hh = h + dy;
        if (hh < 0 || hh >= Himg) continue;
        #pragma unroll
        for (int dx = -1; dx <= 1; ++dx) {
            int ww = w_ + dx;
            if (ww < 0 || ww >= Wimg) continue;
            acc += ib[hh * Wimg + ww] * wc[(dy + 1) * 3 + (dx + 1)];
        }
    }
    out[(size_t)gid] = acc;
}

// ---------- per-(b,c)-row 1/max(||row||,1e-12) ----------
__global__ __launch_bounds__(256) void rowscale(const float* __restrict__ in,
                                                float* __restrict__ scale, int chanPerB) {
    __shared__ float red[256];
    int tid = threadIdx.x;
    int bx = blockIdx.x;                 // b*64+c
    int b = bx >> 6, c = bx & 63;
    const float* rowp = in + (size_t)(b * chanPerB + c) * HW;
    float s = 0.f;
    for (int i = tid; i < HW; i += 256) { float v = rowp[i]; s += v * v; }
    red[tid] = s; __syncthreads();
    for (int off = 128; off; off >>= 1) { if (tid < off) red[tid] += red[tid + off]; __syncthreads(); }
    if (tid == 0) scale[bx] = 1.f / fmaxf(sqrtf(red[0]), 1e-12f);
}

// ---------- 1x1 conv as GEMM: out[Cout,HW] = W[Cout,64] @ in[64,HW]  (bf16 WMMA) ----------
// mode 0: plain; mode 2: out = acc + res
__global__ __launch_bounds__(256) void conv1x1_wmma(const float* __restrict__ in,
                                                    const unsigned short* __restrict__ wbf,
                                                    float* __restrict__ out,
                                                    const float* __restrict__ res,
                                                    int Cout, int mode) {
    __shared__ __align__(16) unsigned short Alds[128 * LSTR];
    __shared__ __align__(16) unsigned short Bt[32 * LSTR];     // transposed: [pixel][k]
    const int tid = threadIdx.x;
    const int wave = tid >> 5, lane = tid & 31;
    const int laneRow = lane & 15, hl = lane >> 4;
    const int wavesM = Cout >> 4;
    const int pixTile = (8 / wavesM) * 16;            // 16 (Cout=128) or 32 (Cout=64)
    const int m0   = (wave % wavesM) * 16;
    const int nOff = (wave / wavesM) * 16;
    const int b = blockIdx.y;
    const int p0 = blockIdx.x * pixTile;

    v8f acc; for (int i = 0; i < 8; ++i) acc[i] = 0.f;

    for (int k0 = 0; k0 < 64; k0 += 32) {
        __syncthreads();
        // A chunk: Cout x 32 (ushort4 vector path)
        for (int j = tid; j < Cout * 8; j += 256) {
            int m = j >> 3, kk = (j & 7) * 4;
            *(ush4*)&Alds[m * LSTR + kk] = *(const ush4*)&wbf[m * 64 + k0 + kk];
        }
        // B chunk (transposed store): 32 x pixTile, float4 global loads
        for (int j = tid; j < (32 * pixTile) / 4; j += 256) {
            int idx = j * 4;
            int kk = idx / pixTile, p = idx % pixTile;
            float4 v = *(const float4*)&in[(size_t)(b * CC + k0 + kk) * HW + p0 + p];
            Bt[(p + 0) * LSTR + kk] = f2b_bits(v.x);
            Bt[(p + 1) * LSTR + kk] = f2b_bits(v.y);
            Bt[(p + 2) * LSTR + kk] = f2b_bits(v.z);
            Bt[(p + 3) * LSTR + kk] = f2b_bits(v.w);
        }
        __syncthreads();
        const unsigned short* ap = &Alds[(m0 + laneRow) * LSTR];
        v16bf a  = frag16(ap + hl * 8, ap + 16 + hl * 8);
        const unsigned short* bp = &Bt[(nOff + laneRow) * LSTR + hl * 8];
        v16bf bm = frag16(bp, bp + 16);
        acc = __builtin_amdgcn_wmma_f32_16x16x32_bf16(false, a, false, bm,
                                                      (short)0, acc, false, false);
    }
    int p = p0 + nOff + laneRow;
    #pragma unroll
    for (int r = 0; r < 8; ++r) {
        int m = m0 + r + hl * 8;
        size_t idx = (size_t)(b * Cout + m) * HW + p;
        float v = acc[r];
        if (mode == 2) v += res[idx];
        out[idx] = v;
    }
}

// ---------- 3x3 conv 64->64 as implicit GEMM (K=576, bf16 WMMA) ----------
// mode 0: plain; mode 1: gelu(acc); mode 2: acc + res
__global__ __launch_bounds__(256) void conv3x3_wmma(const float* __restrict__ in,
                                                    const unsigned short* __restrict__ wbf,
                                                    float* __restrict__ out,
                                                    const float* __restrict__ res,
                                                    int mode) {
    __shared__ __align__(16) unsigned short Alds[64 * LSTR];
    __shared__ __align__(16) unsigned short Bt[32 * LSTR];     // transposed: [pixel][k]
    const int tid = threadIdx.x;
    const int wave = tid >> 5, lane = tid & 31;
    const int laneRow = lane & 15, hl = lane >> 4;
    const int m0 = (wave & 3) * 16;
    const int nOff = (wave >> 2) * 16;
    const int w0 = blockIdx.x * 32;
    const int h0 = blockIdx.y;
    const int b  = blockIdx.z;

    v8f acc; for (int i = 0; i < 8; ++i) acc[i] = 0.f;

    for (int k0 = 0; k0 < 576; k0 += 32) {
        __syncthreads();
        // A chunk: 64 x 32 (ushort4 vector path)
        for (int j = tid; j < 64 * 8; j += 256) {
            int m = j >> 3, kk = (j & 7) * 4;
            *(ush4*)&Alds[m * LSTR + kk] = *(const ush4*)&wbf[m * 576 + k0 + kk];
        }
        if (k0 + 32 < 576)  // prefetch next weight chunk (global_prefetch_b8)
            __builtin_prefetch(&wbf[(m0 + laneRow) * 576 + k0 + 32], 0, 1);
        // B chunk (implicit im2col, transposed store)
        for (int i = tid; i < 32 * 32; i += 256) {
            int kk = i >> 5, p = i & 31;
            int K = k0 + kk;
            int c = K / 9, r9 = K % 9;
            int dy = r9 / 3 - 1, dx = r9 % 3 - 1;
            int hh = h0 + dy, ww = w0 + p + dx;
            float v = 0.f;
            if (hh >= 0 && hh < Himg && ww >= 0 && ww < Wimg)
                v = in[(size_t)(b * CC + c) * HW + hh * Wimg + ww];
            Bt[p * LSTR + kk] = f2b_bits(v);
        }
        __syncthreads();
        const unsigned short* ap = &Alds[(m0 + laneRow) * LSTR];
        v16bf a  = frag16(ap + hl * 8, ap + 16 + hl * 8);
        const unsigned short* bp = &Bt[(nOff + laneRow) * LSTR + hl * 8];
        v16bf bm = frag16(bp, bp + 16);
        acc = __builtin_amdgcn_wmma_f32_16x16x32_bf16(false, a, false, bm,
                                                      (short)0, acc, false, false);
    }
    int p = nOff + laneRow;
    #pragma unroll
    for (int r = 0; r < 8; ++r) {
        int m = m0 + r + hl * 8;
        size_t idx = (size_t)(b * CC + m) * HW + h0 * Wimg + w0 + p;
        float v = acc[r];
        if (mode == 1)      v = gelu_exact(v);
        else if (mode == 2) v += res[idx];
        out[idx] = v;
    }
}

// ---------- fused QK^T (f32 WMMA, K=65536) + scale-fold + softmax ----------
// q: [B,64,HW] (Q2);  kv2: [B,128,HW] (channels 0..63 = k_raw)
__global__ __launch_bounds__(256) void qk_softmax(const float* __restrict__ q,
                                                  const float* __restrict__ kv2,
                                                  const float* __restrict__ qs,
                                                  const float* __restrict__ ks,
                                                  const float* __restrict__ temp,
                                                  float* __restrict__ attn) {
    __shared__ __align__(16) float qt[16 * 128];
    __shared__ __align__(16) float kt[16 * 128];
    __shared__ float part[8 * 256];
    __shared__ float sm[256];
    const int tid = threadIdx.x;
    const int wave = tid >> 5, lane = tid & 31;
    const int laneRow = lane & 15, hl = lane >> 4;
    const int bh = blockIdx.x, b = bh >> 2, head = bh & 3;

    v8f acc; for (int i = 0; i < 8; ++i) acc[i] = 0.f;

    for (int t0 = 0; t0 < HW; t0 += 128) {
        __syncthreads();
        for (int i = tid; i < 16 * 128 / 4; i += 256) {
            int r = i >> 5, cpos = (i & 31) * 4;
            *(float4*)&qt[r * 128 + cpos] =
                *(const float4*)&q  [(size_t)(b * 64  + head * 16 + r) * HW + t0 + cpos];
            *(float4*)&kt[r * 128 + cpos] =
                *(const float4*)&kv2[(size_t)(b * 128 + head * 16 + r) * HW + t0 + cpos];
        }
        __syncthreads();
        int nw = wave * 16;
        #pragma unroll
        for (int kb = 0; kb < 16; kb += 4) {
            v2f a  = *(const v2f*)(qt + laneRow * 128 + nw + kb + hl * 2);
            v2f bm = *(const v2f*)(kt + laneRow * 128 + nw + kb + hl * 2);
            acc = __builtin_amdgcn_wmma_f32_16x16x4_f32(false, a, false, bm,
                                                        (short)0, acc, false, false);
        }
    }
    #pragma unroll
    for (int r = 0; r < 8; ++r)
        part[wave * 256 + (r + hl * 8) * 16 + laneRow] = acc[r];
    __syncthreads();
    {
        int m = tid >> 4, d = tid & 15;
        float s = 0.f;
        #pragma unroll
        for (int w = 0; w < 8; ++w) s += part[w * 256 + tid];
        s *= qs[b * 64 + head * 16 + m] * ks[b * 64 + head * 16 + d] * temp[head];
        sm[tid] = s;
    }
    __syncthreads();
    if (tid < 16) {
        int row = tid;
        float mx = -3.4e38f;
        for (int d = 0; d < 16; ++d) mx = fmaxf(mx, sm[row * 16 + d]);
        float sum = 0.f;
        float e[16];
        for (int d = 0; d < 16; ++d) { e[d] = expf(sm[row * 16 + d] - mx); sum += e[d]; }
        float inv = 1.f / sum;
        for (int d = 0; d < 16; ++d) attn[bh * 256 + row * 16 + d] = e[d] * inv;
    }
}

// ---------- out = attn @ v  (f32 WMMA, M=16,K=16,N=HW) ----------
__global__ __launch_bounds__(256) void attnv_wmma(const float* __restrict__ attn,
                                                  const float* __restrict__ kv2,
                                                  float* __restrict__ ao) {
    __shared__ __align__(16) float At[256];
    const int tid = threadIdx.x;
    const int wave = tid >> 5, lane = tid & 31;
    const int laneRow = lane & 15, hl = lane >> 4;
    const int bh = blockIdx.y, b = bh >> 2, head = bh & 3;
    const int n0 = blockIdx.x * 128 + wave * 16;

    At[tid] = attn[bh * 256 + tid];
    __syncthreads();

    v8f acc; for (int i = 0; i < 8; ++i) acc[i] = 0.f;
    #pragma unroll
    for (int kb = 0; kb < 16; kb += 4) {
        v2f a = *(const v2f*)(At + laneRow * 16 + kb + hl * 2);
        v2f bm;
        #pragma unroll
        for (int j = 0; j < 2; ++j) {
            int d = kb + hl * 2 + j;
            bm[j] = kv2[(size_t)(b * 128 + 64 + head * 16 + d) * HW + n0 + laneRow];
        }
        acc = __builtin_amdgcn_wmma_f32_16x16x4_f32(false, a, false, bm,
                                                    (short)0, acc, false, false);
    }
    #pragma unroll
    for (int r = 0; r < 8; ++r) {
        int m = r + hl * 8;
        ao[(size_t)(b * 64 + head * 16 + m) * HW + n0 + laneRow] = acc[r];
    }
}

// ---------- z = t2 * sigmoid(m2) + xa ----------
__global__ __launch_bounds__(256) void zmix(const float* __restrict__ t2,
                                            const float* __restrict__ m2,
                                            const float* __restrict__ xa,
                                            float* __restrict__ z) {
    size_t i = (size_t)blockIdx.x * 256 + threadIdx.x;
    float s = 1.f / (1.f + expf(-m2[i]));
    z[i] = t2[i] * s + xa[i];
}

// =====================================================================
extern "C" void kernel_launch(void* const* d_in, const int* in_sizes, int n_in,
                              void* d_out, int out_size, void* d_ws, size_t ws_size,
                              hipStream_t stream) {
    (void)in_sizes; (void)n_in; (void)out_size; (void)ws_size;
    const float* x     = (const float*)d_in[0];
    const float* ref   = (const float*)d_in[1];
    const float* ln1w  = (const float*)d_in[2];
    const float* ln1b  = (const float*)d_in[3];
    const float* ln2w  = (const float*)d_in[4];
    const float* ln2b  = (const float*)d_in[5];
    const float* ln3w  = (const float*)d_in[6];
    const float* ln3b  = (const float*)d_in[7];
    const float* wkv   = (const float*)d_in[8];
    const float* wkvdw = (const float*)d_in[9];
    const float* wq    = (const float*)d_in[10];
    const float* wqdw  = (const float*)d_in[11];
    const float* wproj = (const float*)d_in[12];
    const float* temp  = (const float*)d_in[13];
    const float* wadd1 = (const float*)d_in[14];
    const float* wadd2 = (const float*)d_in[15];
    const float* wmul1 = (const float*)d_in[16];
    const float* wfuse = (const float*)d_in[17];
    float* out = (float*)d_out;

    const size_t S = (size_t)BB * CC * HW * sizeof(float);   // 64 MiB
    char* W = (char*)d_ws;
    float* f0 = (float*)(W + 0 * S);
    float* f1 = (float*)(W + 1 * S);
    float* f2 = (float*)(W + 2 * S);   // f2..f3: 2S region for KV / then AO,X1
    float* f3 = (float*)(W + 3 * S);
    float* f4 = (float*)(W + 4 * S);   // f4..f5: 2S region for KV2 (k_raw | v) / then M2
    char* aux = W + 6 * S;
    unsigned short* wb_kv = (unsigned short*)aux;            aux += 8192 * 2;
    unsigned short* wb_q  = (unsigned short*)aux;            aux += 4096 * 2;
    unsigned short* wb_pr = (unsigned short*)aux;            aux += 4096 * 2;
    unsigned short* wb_a1 = (unsigned short*)aux;            aux += 36864 * 2;
    unsigned short* wb_a2 = (unsigned short*)aux;            aux += 36864 * 2;
    unsigned short* wb_m1 = (unsigned short*)aux;            aux += 36864 * 2;
    unsigned short* wb_fz = (unsigned short*)aux;            aux += 36864 * 2;
    float* qsc  = (float*)aux;                               aux += 256 * 4;
    float* ksc  = (float*)aux;                               aux += 256 * 4;
    float* attn = (float*)aux;                               aux += 4096 * 4;

    // 1) prepack weights to bf16
    prep_bf16<<<8192 / 256, 256, 0, stream>>>(wkv,   wb_kv, 8192);
    prep_bf16<<<4096 / 256, 256, 0, stream>>>(wq,    wb_q,  4096);
    prep_bf16<<<4096 / 256, 256, 0, stream>>>(wproj, wb_pr, 4096);
    prep_bf16<<<36864 / 256, 256, 0, stream>>>(wadd1, wb_a1, 36864);
    prep_bf16<<<36864 / 256, 256, 0, stream>>>(wadd2, wb_a2, 36864);
    prep_bf16<<<36864 / 256, 256, 0, stream>>>(wmul1, wb_m1, 36864);
    prep_bf16<<<36864 / 256, 256, 0, stream>>>(wfuse, wb_fz, 36864);

    const int lnBlocks = BB * HW / 256;                      // 1024
    // 2) ln1(ref)->f0 ; ln3(x)->f1
    layernorm_c_kernel<<<lnBlocks, 256, 0, stream>>>(ref, ln1w, ln1b, f0);
    layernorm_c_kernel<<<lnBlocks, 256, 0, stream>>>(x,   ln3w, ln3b, f1);
    // 3) KV = w_kv @ ln1  -> region f2 (2S, [B,128,HW])
    conv1x1_wmma<<<dim3(HW / 16, BB), 256, 0, stream>>>(f0, wb_kv, f2, nullptr, 128, 0);
    // 4) depthwise on KV -> region f4 (2S)
    dwconv3x3<<<(BB * 128 * HW) / 256, 256, 0, stream>>>(f2, wkvdw, f4, 128);
    // 5) Q = w_q @ ln3 -> f0 ; 6) depthwise -> f1
    conv1x1_wmma<<<dim3(HW / 32, BB), 256, 0, stream>>>(f1, wb_q, f0, nullptr, 64, 0);
    dwconv3x3<<<(BB * 64 * HW) / 256, 256, 0, stream>>>(f0, wqdw, f1, 64);
    // 7) l2 row scales for q (f1, [B,64,HW]) and k (f4, [B,128,HW] ch 0..63)
    rowscale<<<BB * 64, 256, 0, stream>>>(f1, qsc, 64);
    rowscale<<<BB * 64, 256, 0, stream>>>(f4, ksc, 128);
    // 8) attn = softmax( (q k^T) * qs*ks*temp )   (f32 WMMA)
    qk_softmax<<<BB * 4, 256, 0, stream>>>(f1, f4, qsc, ksc, temp, attn);
    // 9) AO = attn @ v -> f2   (f32 WMMA)
    attnv_wmma<<<dim3(HW / 128, BB * 4), 256, 0, stream>>>(attn, f4, f2);
    // 10) X1 = x + w_proj @ AO -> f3
    conv1x1_wmma<<<dim3(HW / 32, BB), 256, 0, stream>>>(f2, wb_pr, f3, x, 64, 2);
    // 11) T2 = ln2(X1) -> f0
    layernorm_c_kernel<<<lnBlocks, 256, 0, stream>>>(f3, ln2w, ln2b, f0);
    // 12) G1 = gelu(conv3(T2,w_add1)) -> f1
    conv3x3_wmma<<<dim3(Wimg / 32, Himg, BB), 256, 0, stream>>>(f0, wb_a1, f1, nullptr, 1);
    // 13) XA = conv3(G1,w_add2) -> f2
    conv3x3_wmma<<<dim3(Wimg / 32, Himg, BB), 256, 0, stream>>>(f1, wb_a2, f2, nullptr, 0);
    // 14) G2 = gelu(conv3(T2,w_mul1)) -> f1
    conv3x3_wmma<<<dim3(Wimg / 32, Himg, BB), 256, 0, stream>>>(f0, wb_m1, f1, nullptr, 1);
    // 15) M2 = conv3(G2,w_mul1) -> f4
    conv3x3_wmma<<<dim3(Wimg / 32, Himg, BB), 256, 0, stream>>>(f1, wb_m1, f4, nullptr, 0);
    // 16) Z = T2*sigmoid(M2)+XA -> f1
    zmix<<<(BB * CC * HW) / 256, 256, 0, stream>>>(f0, f4, f2, f1);
    // 17) out = X1 + conv3(Z,w_fuse)
    conv3x3_wmma<<<dim3(Wimg / 32, Himg, BB), 256, 0, stream>>>(f1, wb_fz, out, f3, 2);
}
